// AttentionBase_72636486910509
// MI455X (gfx1250) — compile-verified, matrix operand order checked by reading, not run on hardware
//
#include <hip/hip_runtime.h>

// Problem constants (match reference)
#define B_   2
#define N_   2048
#define H_   16
#define D_   64
#define MODEL_ 1024

typedef __attribute__((ext_vector_type(16))) __bf16 v16bf;
typedef __attribute__((ext_vector_type(8)))  __bf16 v8bf;
typedef __attribute__((ext_vector_type(8)))  float  v8f;
typedef int v4i __attribute__((__vector_size__(16)));   // matches builtin param type

#define AS1 __attribute__((address_space(1)))
#define AS3 __attribute__((address_space(3)))

// ---- CDNA5 async global->LDS staging (feature-detected; fallback = sync path)
#if defined(__has_builtin)
#if __has_builtin(__builtin_amdgcn_global_load_async_to_lds_b128)
#define HAVE_ASYNC 1
#endif
#endif

#ifdef HAVE_ASYNC
static __device__ inline void async_cp16(const float* g, float* l) {
  // copies 16B per lane: LDS[l] = GLOBAL[g]; tracked by ASYNCcnt
  __builtin_amdgcn_global_load_async_to_lds_b128(
      (AS1 v4i*)const_cast<float*>(g), (AS3 v4i*)l, 0, 0);
}
#if __has_builtin(__builtin_amdgcn_s_wait_asynccnt)
#define WAIT_ASYNC(n) do { __builtin_amdgcn_s_wait_asynccnt(n); asm volatile("" ::: "memory"); } while (0)
#else
#define WAIT_ASYNC(n) asm volatile("s_wait_asynccnt %0" :: "i"(n) : "memory")
#endif
#endif

static __device__ inline v8f wmma_bf16(v16bf a, v16bf b, v8f c) {
  // (neg_a, A, neg_b, B, c_mod, C, reuse_a, reuse_b)
  return __builtin_amdgcn_wmma_f32_16x16x32_bf16(false, a, false, b, (short)0, c,
                                                 false, false);
}

static __device__ inline v8f v8f_zero() {
  v8f r;
#pragma unroll
  for (int i = 0; i < 8; ++i) r[i] = 0.0f;
  return r;
}

// Load a 16-element bf16 fragment (A or B layout) from row-major fp32 memory.
// Chunk0: cols [dd + 8*half, +8), Chunk1: cols [dd + 16 + 8*half, +8).
static __device__ inline v16bf load_frag_g(const float* __restrict__ rowp, int dd, int half) {
  const v8f c0 = *(const v8f*)(rowp + dd + 8 * half);
  const v8f c1 = *(const v8f*)(rowp + dd + 16 + 8 * half);
  v16bf r;
#pragma unroll
  for (int i = 0; i < 8; ++i) {
    r[i]     = (__bf16)c0[i];
    r[8 + i] = (__bf16)c1[i];
  }
  return r;
}

// Same fragment pattern from bf16 LDS (two ds_load_b128).
static __device__ inline v16bf load_frag_lds(const __bf16* rowp, int dd, int half) {
  const v8bf c0 = *(const v8bf*)(rowp + dd + 8 * half);
  const v8bf c1 = *(const v8bf*)(rowp + dd + 16 + 8 * half);
  v16bf r;
#pragma unroll
  for (int i = 0; i < 8; ++i) {
    r[i]     = c0[i];
    r[8 + i] = c1[i];
  }
  return r;
}

// ---------------------------------------------------------------------------
// Kernel 1: flash attention with T5 relative-position bias.
// Grid: B*H*(N/128) blocks of 256 threads (8 wave32). Block owns 128 queries
// of one (b,h); wave owns 16 queries. Writes fp32 attn output [B*N, MODEL].
// ---------------------------------------------------------------------------
#define KSTRIDE 72     // padded bf16 row stride for K tile   (144B, 16B-multiple)
#define VSTRIDE 40     // padded bf16 row stride for VT tile  (80B, 16B-multiple)
#define BIAS_SZ 2304   // rel window for one 128-query block: idx = key - wid*16 - ln + 127

__global__ __launch_bounds__(256) void attn_fwd_kernel(
    const float* __restrict__ Qg, const float* __restrict__ Kg,
    const float* __restrict__ Vg, const float* __restrict__ Rel,
    float* __restrict__ Og) {
  __shared__ float s_bias[BIAS_SZ];                      // scaled Toeplitz bias
  __shared__ alignas(16) __bf16 s_k[32 * KSTRIDE];       // K tile  [32 keys][64 d]
  __shared__ alignas(16) __bf16 s_vt[64 * VSTRIDE];      // V tile transposed [64 d][32 keys]
#ifdef HAVE_ASYNC
  __shared__ alignas(16) float s_f32[2][4096];           // double-buffered raw K|V fp32 tiles
#endif

  const int tid  = threadIdx.x;
  const int lane = tid & 31;
  const int wid  = tid >> 5;
  const int half = lane >> 4;
  const int ln   = lane & 15;

  const int bm = blockIdx.x & 15;          // query block (N/128 = 16)
  const int h  = (blockIdx.x >> 4) & 15;
  const int b  = blockIdx.x >> 8;

  const int srow = tid >> 3;            // staging: row 0..31
  const int scol = (tid & 7) * 8;       // staging: col chunk 0..56
  const float* gK0 = Kg + ((size_t)b * N_ + srow) * MODEL_ + h * D_ + scol;
  const float* gV0 = Vg + ((size_t)b * N_ + srow) * MODEL_ + h * D_ + scol;

#ifdef HAVE_ASYNC
  // prologue: async-prefetch tile j=0 into buffer 0
  {
    float* lk = &s_f32[0][srow * 64 + scol];
    float* lv = &s_f32[0][2048 + srow * 64 + scol];
    async_cp16(gK0, lk);     async_cp16(gK0 + 4, lk + 4);
    async_cp16(gV0, lv);     async_cp16(gV0 + 4, lv + 4);
  }
  int buf = 0;
#endif

  // ---- Precompute scaled rel-pos bias table for this block's rel window.
  // rel = idx - bm*128 - 127;  bias = rel_emb[bucket(rel)][h] * D^-0.5
  const float inv_log16_x8 = 8.0f / __logf(16.0f);
  for (int idx = tid; idx < BIAS_SZ; idx += 256) {
    int rel = idx - bm * 128 - 127;
    int ret = (rel >= 0) ? 16 : 0;
    int n   = (rel < 0) ? -rel : rel;
    int bucket;
    if (n < 8) {
      bucket = ret + n;
    } else {
      int vl = 8 + (int)(__logf((float)n * 0.125f) * inv_log16_x8);
      vl = min(vl, 15);
      bucket = ret + vl;
    }
    s_bias[idx] = Rel[bucket * H_ + h] * 0.125f;  // fold D^-0.5 scale
  }

  // ---- Q fragments held in registers for the whole kernel (B-matrix layout;
  //      lane = query column, VGPRs = d contraction values).
  const int qbase = bm * 128 + wid * 16;
  const int qg    = qbase + ln;  // this lane's query position
  const float* qrow = Qg + ((size_t)b * N_ + qg) * MODEL_ + h * D_;
  const v16bf qf0 = load_frag_g(qrow, 0, half);
  const v16bf qf1 = load_frag_g(qrow, 32, half);

  v8f oacc[4];
#pragma unroll
  for (int g = 0; g < 4; ++g) oacc[g] = v8f_zero();
  float m_prev = -1e30f, lsum = 0.0f;

  for (int j = 0; j < N_; j += 32) {
#ifdef HAVE_ASYNC
    // issue next tile's async copy, then wait for current tile's 4 copies
    if (j + 32 < N_) {
      const float* gk = gK0 + (size_t)(j + 32) * MODEL_;
      const float* gv = gV0 + (size_t)(j + 32) * MODEL_;
      float* lk = &s_f32[buf ^ 1][srow * 64 + scol];
      float* lv = &s_f32[buf ^ 1][2048 + srow * 64 + scol];
      async_cp16(gk, lk);  async_cp16(gk + 4, lk + 4);
      async_cp16(gv, lv);  async_cp16(gv + 4, lv + 4);
      WAIT_ASYNC(4);   // in-order completion: current tile's copies are done
    } else {
      WAIT_ASYNC(0);
    }
#endif
    __syncthreads();   // all waves done reading previous s_k / s_vt (also covers bias init)
    // ---- convert own fp32 chunk -> bf16 K tile (row-major) + VT tile (transposed)
    {
#ifdef HAVE_ASYNC
      const v8f kv = *(const v8f*)&s_f32[buf][srow * 64 + scol];
      const v8f vv = *(const v8f*)&s_f32[buf][2048 + srow * 64 + scol];
#else
      const v8f kv = *(const v8f*)(gK0 + (size_t)j * MODEL_);
      const v8f vv = *(const v8f*)(gV0 + (size_t)j * MODEL_);
#endif
      v8bf kb;
#pragma unroll
      for (int i = 0; i < 8; ++i) kb[i] = (__bf16)kv[i];
      *(v8bf*)&s_k[srow * KSTRIDE + scol] = kb;
#pragma unroll
      for (int i = 0; i < 8; ++i) s_vt[(scol + i) * VSTRIDE + srow] = (__bf16)vv[i];
    }
    __syncthreads();

    // ---- S^T = K_tile x Q^T : two 16x16 tiles (keys 0..15 and 16..31)
    v8f c0 = v8f_zero(), c1 = v8f_zero();
    {
      const __bf16* kr = &s_k[ln * KSTRIDE];
      c0 = wmma_bf16(load_frag_lds(kr, 0, half), qf0, c0);
      c0 = wmma_bf16(load_frag_lds(kr, 32, half), qf1, c0);
    }
    {
      const __bf16* kr = &s_k[(16 + ln) * KSTRIDE];
      c1 = wmma_bf16(load_frag_lds(kr, 0, half), qf0, c1);
      c1 = wmma_bf16(load_frag_lds(kr, 32, half), qf1, c1);
    }

    // ---- bias + scale; online softmax over the 16 keys this lane holds.
    // C-layout: VGPR r = key (j + 8*half + r) [tile0] / +16 [tile1]; lane = query.
    float s0[8], s1[8];
    const int bidx = j + 8 * half - wid * 16 - ln + 127;
    float mloc = -1e30f;
#pragma unroll
    for (int r = 0; r < 8; ++r) {
      s0[r] = c0[r] * 0.125f + s_bias[bidx + r];
      s1[r] = c1[r] * 0.125f + s_bias[bidx + 16 + r];
      mloc  = fmaxf(mloc, fmaxf(s0[r], s1[r]));
    }
    mloc = fmaxf(mloc, __shfl_xor(mloc, 16, 32));
    const float m_new = fmaxf(m_prev, mloc);
    const float alpha = __expf(m_prev - m_new);
    float psum = 0.0f;
#pragma unroll
    for (int r = 0; r < 8; ++r) {
      s0[r] = __expf(s0[r] - m_new);
      s1[r] = __expf(s1[r] - m_new);
      psum += s0[r] + s1[r];
    }
    psum += __shfl_xor(psum, 16, 32);
    lsum = lsum * alpha + psum;
    m_prev = m_new;

    // ---- rescale O by alpha of its row (query r+8*half lives in lane r+8*half)
#pragma unroll
    for (int r = 0; r < 8; ++r) {
      const float ab = __shfl(alpha, r + 8 * half, 32);
#pragma unroll
      for (int g = 0; g < 4; ++g) oacc[g][r] *= ab;
    }

    // ---- P in C-layout is already the A-fragment layout: pack in-lane.
    v16bf pf;
#pragma unroll
    for (int r = 0; r < 8; ++r) {
      pf[r]     = (__bf16)s0[r];
      pf[8 + r] = (__bf16)s1[r];
    }
    // ---- O += P x V  (B-fragment from transposed V tile: contiguous keys)
#pragma unroll
    for (int g = 0; g < 4; ++g) {
      const v16bf vf = load_frag_lds(&s_vt[(g * 16 + ln) * VSTRIDE], 0, half);
      oacc[g] = wmma_bf16(pf, vf, oacc[g]);
    }
#ifdef HAVE_ASYNC
    buf ^= 1;
#endif
  }

  // ---- normalize and store fp32 intermediate [B*N, MODEL]
  const float linv = 1.0f / lsum;
#pragma unroll
  for (int r = 0; r < 8; ++r) {
    const float lb = __shfl(linv, r + 8 * half, 32);
    const size_t row = (size_t)b * N_ + qbase + r + 8 * half;
#pragma unroll
    for (int g = 0; g < 4; ++g)
      Og[row * MODEL_ + h * D_ + g * 16 + ln] = oacc[g][r] * lb;
  }
}

// ---------------------------------------------------------------------------
// Kernel 2: out = X @ W^T + b.  X:[4096,1024] fp32 (ws), W:[1024,1024].
// Per wave: one 16x64 tile (A-fragment reused across 4 B-fragments).
// ---------------------------------------------------------------------------
__global__ __launch_bounds__(256) void proj_kernel(
    const float* __restrict__ X, const float* __restrict__ W,
    const float* __restrict__ bias, float* __restrict__ out) {
  const int tid  = threadIdx.x;
  const int lane = tid & 31;
  const int wid  = tid >> 5;
  const int half = lane >> 4;
  const int ln   = lane & 15;

  const int job = blockIdx.x * 8 + wid;  // 4096 wave jobs
  const int mt  = job >> 4;              // 256 row tiles
  const int ng  = job & 15;              // 16 groups of 64 cols

  const int arow = mt * 16 + ln;
  v8f acc[4];
#pragma unroll
  for (int t = 0; t < 4; ++t) acc[t] = v8f_zero();

  for (int kk = 0; kk < MODEL_; kk += 32) {
    const v16bf af = load_frag_g(X + (size_t)arow * MODEL_ + kk, 0, half);
#pragma unroll
    for (int t = 0; t < 4; ++t) {
      const int col = ng * 64 + t * 16 + ln;
      const v16bf bf = load_frag_g(W + (size_t)col * MODEL_ + kk, 0, half);
      acc[t] = wmma_bf16(af, bf, acc[t]);
    }
  }

#pragma unroll
  for (int t = 0; t < 4; ++t) {
    const int col = ng * 64 + t * 16 + ln;
    const float bo = bias[col];
#pragma unroll
    for (int r = 0; r < 8; ++r)
      out[(size_t)(mt * 16 + r + 8 * half) * MODEL_ + col] = acc[t][r] + bo;
  }
}

// ---------------------------------------------------------------------------
extern "C" void kernel_launch(void* const* d_in, const int* in_sizes, int n_in,
                              void* d_out, int out_size, void* d_ws, size_t ws_size,
                              hipStream_t stream) {
  const float* q   = (const float*)d_in[0];
  const float* k   = (const float*)d_in[1];
  const float* v   = (const float*)d_in[2];
  const float* rel = (const float*)d_in[3];
  const float* w   = (const float*)d_in[4];
  const float* bo  = (const float*)d_in[5];
  float* out  = (float*)d_out;
  float* attn = (float*)d_ws;  // needs B*N*MODEL*4 = 16 MiB scratch

  attn_fwd_kernel<<<dim3(B_ * H_ * (N_ / 128)), 256, 0, stream>>>(q, k, v, rel, attn);
  proj_kernel<<<dim3((B_ * N_ / 16) * (MODEL_ / 64) / 8), 256, 0, stream>>>(attn, w, bo, out);
}
